// SinkhornAttention_89000312308060
// MI455X (gfx1250) — matrix operand order repaired and checked by reading, not compiled, not run on hardware
//
#include <hip/hip_runtime.h>
#include <math.h>

// Problem constants (match reference)
#define T_TOK 2048
#define N_PAR 2048
#define BATCH 8
#define DM    512
#define SINK_ITERS 5
// SINKHORN_REG == 1.0, so cost / reg == cost

typedef __attribute__((ext_vector_type(16))) __bf16 v16bf;
typedef __attribute__((ext_vector_type(8)))  float  v8f;

// 32-byte WMMA operand fragment viewed as two b128 chunks
union Frag16 { v16bf v; uint4 q[2]; };
// 8 packed bf16 for b128 stores
union Pack8 { uint4 q; __bf16 e[8]; };

// LDS macro-tile geometry for the big GEMMs
#define MROWS 128              // macro tile extent (rows of A / cols of B)
#define ROWE  40               // padded LDS row length in bf16 elems (32 data + 8 pad)
                               // 80B row stride = 20 banks -> conflict-free b128 reads

// ---------- helpers ----------

__device__ inline __bf16 f2bf(float f) {
  // native f32 -> bf16 convert (v_cvt_pk_bf16_f32 on gfx1250)
  return (__bf16)f;
}

__device__ inline v8f wmma_bf16(v16bf a, v16bf b, v8f c) {
  return __builtin_amdgcn_wmma_f32_16x16x32_bf16(false, a, false, b,
                                                 (short)0, c, false, false);
}

// CDNA5 async copy: global -> LDS, 16B per lane, tracked by ASYNCcnt.
// Low 32 bits of a flat LDS pointer are the wave-relative LDS offset (ISA 10.2).
__device__ inline void async_load_b128(const void* g, const void* lds_dst) {
  const unsigned loff = (unsigned)(size_t)lds_dst;
  asm volatile("global_load_async_to_lds_b128 %0, %1, off"
               :: "v"(loff), "v"(g) : "memory");
}
__device__ inline void wait_async4() { asm volatile("s_wait_asynccnt 0x4" ::: "memory"); }
__device__ inline void wait_async0() { asm volatile("s_wait_asynccnt 0x0" ::: "memory"); }

// ---------- kernel 0: weight transpose+convert  Wt[n][k] = bf16(W[k][n]) ----------

__global__ void transpose_w(const float* __restrict__ W, __bf16* __restrict__ Wt)
{
  const int idx = (blockIdx.x * blockDim.x + threadIdx.x) * 8;
  const int n = idx / DM;
  const int k = idx % DM;   // k..k+7 contiguous in the output row
  Pack8 p;
#pragma unroll
  for (int i = 0; i < 8; ++i) p.e[i] = f2bf(W[(size_t)(k + i) * DM + n]);
  *(uint4*)(Wt + (size_t)n * DM + k) = p.q;
}

// ---------- kernel 1: projection GEMM  C[M,512] = A[M,512] x W[512,512] ----------
// One wave computes a 16x64 tile (4 WMMAs / K-step, A fragment reused).

__global__ void proj_gemm(const float* __restrict__ A, const __bf16* __restrict__ Wt,
                          float* __restrict__ C, __bf16* __restrict__ Cbf,
                          __bf16* __restrict__ CbfT, int M)
{
  const int lane = threadIdx.x & 31;
  const int wave = threadIdx.x >> 5;
  const int tiles_n = DM / 64;                       // 8 supertiles across columns
  const int tile = blockIdx.x * (blockDim.x >> 5) + wave;
  const int tm = tile / tiles_n;
  const int tn = tile % tiles_n;
  if (tm * 16 >= M) return;

  const int h   = lane >> 4;      // lane half
  const int l16 = lane & 15;
  const int m   = tm * 16 + l16;  // A row owned by this lane

  v8f acc[4] = {};
  for (int k0 = 0; k0 < DM; k0 += 32) {
    Frag16 a;
    const float* Ap = A + (size_t)m * DM + k0 + 8 * h;
    const float4 f0 = *(const float4*)(Ap);
    const float4 f1 = *(const float4*)(Ap + 4);
    const float4 f2 = *(const float4*)(Ap + 16);
    const float4 f3 = *(const float4*)(Ap + 20);
    a.v[0] = f2bf(f0.x); a.v[1] = f2bf(f0.y); a.v[2]  = f2bf(f0.z); a.v[3]  = f2bf(f0.w);
    a.v[4] = f2bf(f1.x); a.v[5] = f2bf(f1.y); a.v[6]  = f2bf(f1.z); a.v[7]  = f2bf(f1.w);
    a.v[8] = f2bf(f2.x); a.v[9] = f2bf(f2.y); a.v[10] = f2bf(f2.z); a.v[11] = f2bf(f2.w);
    a.v[12] = f2bf(f3.x); a.v[13] = f2bf(f3.y); a.v[14] = f2bf(f3.z); a.v[15] = f2bf(f3.w);
#pragma unroll
    for (int s = 0; s < 4; ++s) {
      const int n = tn * 64 + s * 16 + l16;
      const __bf16* Bp = Wt + (size_t)n * DM + k0 + 8 * h;
      Frag16 bm;
      bm.q[0] = *(const uint4*)(Bp);
      bm.q[1] = *(const uint4*)(Bp + 16);
      acc[s] = wmma_bf16(a.v, bm.v, acc[s]);
    }
  }

#pragma unroll
  for (int s = 0; s < 4; ++s) {
    const int n = tn * 64 + s * 16 + l16;
    if (C) {
#pragma unroll
      for (int r = 0; r < 8; ++r)
        C[(size_t)(tm * 16 + r + 8 * h) * DM + n] = acc[s][r];
    }
    if (Cbf) {
#pragma unroll
      for (int r = 0; r < 8; ++r)
        Cbf[(size_t)(tm * 16 + r + 8 * h) * DM + n] = f2bf(acc[s][r]);
    }
    if (CbfT) {
      Pack8 p;
#pragma unroll
      for (int r = 0; r < 8; ++r) p.e[r] = f2bf(acc[s][r]);
      *(uint4*)(CbfT + (size_t)n * M + tm * 16 + 8 * h) = p.q;
    }
  }
}

// ---------- kernel 2: row sum-of-squares ----------

__global__ void row_sumsq(const float* __restrict__ X, float* __restrict__ out, int rows)
{
  const int lane = threadIdx.x & 31;
  const int row  = blockIdx.x * (blockDim.x >> 5) + (threadIdx.x >> 5);
  if (row >= rows) return;
  float s = 0.f;
#pragma unroll
  for (int i = 0; i < DM / 128; ++i) {
    const float4 xv = *(const float4*)(X + (size_t)row * DM + i * 128 + lane * 4);
    s += xv.x * xv.x + xv.y * xv.y + xv.z * xv.z + xv.w * xv.w;
  }
#pragma unroll
  for (int off = 16; off; off >>= 1) s += __shfl_xor(s, off, 32);
  if (lane == 0) out[row] = s;
}

// ---------- kernel 3: cost GEMM + epilogue (LDS macro-tiled, async double-buffer) ----
// Kmat[b,t,n] = -sqrt(max(q2[t] + k2[b,n] - 2 * qp.kp, 0))   (reg == 1)
// Block = 8 waves, 128x128 macro tile; wave w computes rows [w*16, w*16+16) x 128 cols.

__global__ __launch_bounds__(256)
void cost_gemm(const __bf16* __restrict__ qpbf, const __bf16* __restrict__ kpbf,
               const float* __restrict__ q2, const float* __restrict__ k2,
               float* __restrict__ Kmat)
{
  __shared__ __bf16 shA[2][MROWS * ROWE];
  __shared__ __bf16 shB[2][MROWS * ROWE];

  const int b    = blockIdx.y;
  const int tid  = threadIdx.x;
  const int lane = tid & 31;
  const int w    = tid >> 5;
  const int tiles_n = N_PAR / MROWS;                 // 16
  const int tt0 = (blockIdx.x / tiles_n) * MROWS;
  const int tn0 = (blockIdx.x % tiles_n) * MROWS;
  const int h = lane >> 4, l16 = lane & 15;

  const __bf16* Ab = qpbf;
  const __bf16* Bb = kpbf + (size_t)b * N_PAR * DM;

  // staging: 128 rows x 4 chunks(8 elems) per tile = 512 tasks; 2 per thread per tile
  const int r0 = tid >> 2,          c0 = tid & 3;
  const int r1 = (tid + 256) >> 2,  c1 = (tid + 256) & 3;

  v8f acc[8] = {};

  // prologue: stage K-step 0 into buffer 0
  {
    async_load_b128(Ab + (size_t)(tt0 + r0) * DM + c0 * 8, &shA[0][r0 * ROWE + c0 * 8]);
    async_load_b128(Ab + (size_t)(tt0 + r1) * DM + c1 * 8, &shA[0][r1 * ROWE + c1 * 8]);
    async_load_b128(Bb + (size_t)(tn0 + r0) * DM + c0 * 8, &shB[0][r0 * ROWE + c0 * 8]);
    async_load_b128(Bb + (size_t)(tn0 + r1) * DM + c1 * 8, &shB[0][r1 * ROWE + c1 * 8]);
  }

  int bi = 0;
  for (int k0 = 0; k0 < DM; k0 += 32, bi ^= 1) {
    if (k0 + 32 < DM) {
      const int kn = k0 + 32, nb = bi ^ 1;
      async_load_b128(Ab + (size_t)(tt0 + r0) * DM + kn + c0 * 8, &shA[nb][r0 * ROWE + c0 * 8]);
      async_load_b128(Ab + (size_t)(tt0 + r1) * DM + kn + c1 * 8, &shA[nb][r1 * ROWE + c1 * 8]);
      async_load_b128(Bb + (size_t)(tn0 + r0) * DM + kn + c0 * 8, &shB[nb][r0 * ROWE + c0 * 8]);
      async_load_b128(Bb + (size_t)(tn0 + r1) * DM + kn + c1 * 8, &shB[nb][r1 * ROWE + c1 * 8]);
      wait_async4();          // the 4 oldest (current buffer) have completed
    } else {
      wait_async0();
    }
    __syncthreads();          // all waves' staging visible

    Frag16 a;
    const __bf16* Ar = &shA[bi][(w * 16 + l16) * ROWE];
    a.q[0] = *(const uint4*)(Ar + 8 * h);
    a.q[1] = *(const uint4*)(Ar + 16 + 8 * h);
#pragma unroll
    for (int s = 0; s < 8; ++s) {
      const __bf16* Br = &shB[bi][(s * 16 + l16) * ROWE + 8 * h];
      Frag16 bm;
      bm.q[0] = *(const uint4*)(Br);
      bm.q[1] = *(const uint4*)(Br + 16);
      acc[s] = wmma_bf16(a.v, bm.v, acc[s]);
    }
    __syncthreads();          // done reading before buffer is overwritten
  }

  float* Kb = Kmat + (size_t)b * T_TOK * N_PAR;
#pragma unroll
  for (int s = 0; s < 8; ++s) {
    const int n = tn0 + s * 16 + l16;
    const float k2n = k2[(size_t)b * N_PAR + n];
#pragma unroll
    for (int r = 0; r < 8; ++r) {
      const int mr = tt0 + w * 16 + r + 8 * h;
      const float sq = q2[mr] + k2n - 2.0f * acc[s][r];
      Kb[(size_t)mr * N_PAR + n] = -sqrtf(fmaxf(sq, 0.f));
    }
  }
}

// ---------- kernel 4: Sinkhorn u-update (column logsumexp over t) ----------

__global__ void update_u(const float* __restrict__ Kmat, const float* __restrict__ vvec,
                         float* __restrict__ u)
{
  const int b = blockIdx.y;
  const int n = blockIdx.x * blockDim.x + threadIdx.x;
  const float* Kb = Kmat + (size_t)b * T_TOK * N_PAR;
  const float* vb = vvec + (size_t)b * T_TOK;
  float m = -INFINITY, s = 0.f;
  for (int t = 0; t < T_TOK; ++t) {
    const float x = Kb[(size_t)t * N_PAR + n] + vb[t];
    if (x > m) { s = s * expf(m - x) + 1.f; m = x; }
    else       { s += expf(x - m); }
  }
  const float log_a = -logf((float)N_PAR);
  u[(size_t)b * N_PAR + n] = log_a - (m + logf(s));
}

// ---------- kernel 5: Sinkhorn v-update (row logsumexp over n) ----------

__global__ void update_v(const float* __restrict__ Kmat, const float* __restrict__ u,
                         float* __restrict__ vvec)
{
  const int b    = blockIdx.y;
  const int lane = threadIdx.x & 31;
  const int t    = blockIdx.x * (blockDim.x >> 5) + (threadIdx.x >> 5);
  const float* Krow = Kmat + ((size_t)b * T_TOK + t) * N_PAR;
  const float* ub   = u + (size_t)b * N_PAR;
  float m = -INFINITY, s = 0.f;
  for (int n = lane; n < N_PAR; n += 32) {
    const float x = Krow[n] + ub[n];
    if (x > m) { s = s * expf(m - x) + 1.f; m = x; }
    else       { s += expf(x - m); }
  }
#pragma unroll
  for (int off = 16; off; off >>= 1) {
    const float om = __shfl_xor(m, off, 32);
    const float os = __shfl_xor(s, off, 32);
    const float M  = fmaxf(m, om);
    s = s * expf(m - M) + os * expf(om - M);
    m = M;
  }
  if (lane == 0) vvec[(size_t)b * T_TOK + t] = -(m + logf(s));
}

// ---------- kernel 6: materialize pi (bf16) = exp(Kmat + u + v), 8 elems/thread ----------

__global__ void make_pi(const float* __restrict__ Kmat, const float* __restrict__ u,
                        const float* __restrict__ vvec, __bf16* __restrict__ pibf)
{
  const size_t i8 = ((size_t)blockIdx.x * blockDim.x + threadIdx.x) * 8;
  const int n = (int)(i8 % N_PAR);
  const int t = (int)((i8 / N_PAR) % T_TOK);
  const int b = (int)(i8 / ((size_t)N_PAR * T_TOK));
  const float4 k0 = *(const float4*)(Kmat + i8);
  const float4 k1 = *(const float4*)(Kmat + i8 + 4);
  const float4 u0 = *(const float4*)(u + (size_t)b * N_PAR + n);
  const float4 u1 = *(const float4*)(u + (size_t)b * N_PAR + n + 4);
  const float vt = vvec[(size_t)b * T_TOK + t];
  Pack8 p;
  p.e[0] = f2bf(expf(k0.x + u0.x + vt));
  p.e[1] = f2bf(expf(k0.y + u0.y + vt));
  p.e[2] = f2bf(expf(k0.z + u0.z + vt));
  p.e[3] = f2bf(expf(k0.w + u0.w + vt));
  p.e[4] = f2bf(expf(k1.x + u1.x + vt));
  p.e[5] = f2bf(expf(k1.y + u1.y + vt));
  p.e[6] = f2bf(expf(k1.z + u1.z + vt));
  p.e[7] = f2bf(expf(k1.w + u1.w + vt));
  *(uint4*)(pibf + i8) = p.q;
}

// ---------- kernel 7: output GEMM (LDS macro-tiled, async double-buffer) ----------
// out[b,t,d] = sum_n pi[b,t,n] * vp[b,n,d];  vpT[d][b*N+n] bf16.
// Block = 8 waves, 128(t) x 128(d) macro tile, K = n loop (2048, step 32).

__global__ __launch_bounds__(256)
void out_gemm(const __bf16* __restrict__ pibf, const __bf16* __restrict__ vpT,
              float* __restrict__ out)
{
  __shared__ __bf16 shA[2][MROWS * ROWE];
  __shared__ __bf16 shB[2][MROWS * ROWE];

  const int b    = blockIdx.y;
  const int tid  = threadIdx.x;
  const int lane = tid & 31;
  const int w    = tid >> 5;
  const int tiles_d = DM / MROWS;                    // 4
  const int tt0 = (blockIdx.x / tiles_d) * MROWS;
  const int td0 = (blockIdx.x % tiles_d) * MROWS;
  const int h = lane >> 4, l16 = lane & 15;
  const size_t BN = (size_t)BATCH * N_PAR;

  const __bf16* Ab = pibf + (size_t)b * T_TOK * N_PAR;   // row stride N_PAR
  const __bf16* Bb = vpT + (size_t)b * N_PAR;            // row d: stride BN

  const int r0 = tid >> 2,          c0 = tid & 3;
  const int r1 = (tid + 256) >> 2,  c1 = (tid + 256) & 3;

  v8f acc[8] = {};

  {
    async_load_b128(Ab + (size_t)(tt0 + r0) * N_PAR + c0 * 8, &shA[0][r0 * ROWE + c0 * 8]);
    async_load_b128(Ab + (size_t)(tt0 + r1) * N_PAR + c1 * 8, &shA[0][r1 * ROWE + c1 * 8]);
    async_load_b128(Bb + (size_t)(td0 + r0) * BN + c0 * 8, &shB[0][r0 * ROWE + c0 * 8]);
    async_load_b128(Bb + (size_t)(td0 + r1) * BN + c1 * 8, &shB[0][r1 * ROWE + c1 * 8]);
  }

  int bi = 0;
  for (int k0 = 0; k0 < N_PAR; k0 += 32, bi ^= 1) {
    if (k0 + 32 < N_PAR) {
      const int kn = k0 + 32, nb = bi ^ 1;
      async_load_b128(Ab + (size_t)(tt0 + r0) * N_PAR + kn + c0 * 8, &shA[nb][r0 * ROWE + c0 * 8]);
      async_load_b128(Ab + (size_t)(tt0 + r1) * N_PAR + kn + c1 * 8, &shA[nb][r1 * ROWE + c1 * 8]);
      async_load_b128(Bb + (size_t)(td0 + r0) * BN + kn + c0 * 8, &shB[nb][r0 * ROWE + c0 * 8]);
      async_load_b128(Bb + (size_t)(td0 + r1) * BN + kn + c1 * 8, &shB[nb][r1 * ROWE + c1 * 8]);
      wait_async4();
    } else {
      wait_async0();
    }
    __syncthreads();

    Frag16 a;
    const __bf16* Ar = &shA[bi][(w * 16 + l16) * ROWE];
    a.q[0] = *(const uint4*)(Ar + 8 * h);
    a.q[1] = *(const uint4*)(Ar + 16 + 8 * h);
#pragma unroll
    for (int s = 0; s < 8; ++s) {
      const __bf16* Br = &shB[bi][(s * 16 + l16) * ROWE + 8 * h];
      Frag16 bm;
      bm.q[0] = *(const uint4*)(Br);
      bm.q[1] = *(const uint4*)(Br + 16);
      acc[s] = wmma_bf16(a.v, bm.v, acc[s]);
    }
    __syncthreads();
  }

  float* ob = out + (size_t)b * T_TOK * DM;
#pragma unroll
  for (int s = 0; s < 8; ++s) {
    const int d = td0 + s * 16 + l16;
#pragma unroll
    for (int r = 0; r < 8; ++r) {
      const int mr = tt0 + w * 16 + r + 8 * h;
      ob[(size_t)mr * DM + d] = acc[s][r];
    }
  }
}

// ---------- host launcher ----------

extern "C" void kernel_launch(void* const* d_in, const int* in_sizes, int n_in,
                              void* d_out, int out_size, void* d_ws, size_t ws_size,
                              hipStream_t stream)
{
  (void)in_sizes; (void)n_in; (void)out_size; (void)ws_size;

  const float* q  = (const float*)d_in[0];   // [T, DM]
  const float* k  = (const float*)d_in[1];   // [B, N, DM]
  const float* v  = (const float*)d_in[2];   // [B, N, DM]
  const float* Qw = (const float*)d_in[3];   // [DM, DM]
  const float* Kw = (const float*)d_in[4];   // [DM, DM]
  const float* Vw = (const float*)d_in[5];   // [DM, DM]
  float* out = (float*)d_out;                // [B, T, DM]

  char* ws = (char*)d_ws;
  size_t off = 0;
  auto alloc = [&](size_t bytes) -> void* {
    void* p = ws + off;
    off = (off + bytes + 255) & ~(size_t)255;
    return p;
  };

  const size_t BN = (size_t)BATCH * N_PAR;

  float*  qp     = (float*) alloc((size_t)T_TOK * DM * 4);
  float*  kp     = (float*) alloc(BN * DM * 4);
  __bf16* qp_bf  = (__bf16*)alloc((size_t)T_TOK * DM * 2);
  __bf16* kp_bf  = (__bf16*)alloc(BN * DM * 2);
  __bf16* vpT_bf = (__bf16*)alloc((size_t)DM * BN * 2);   // transposed: [DM][B*N]
  __bf16* QwT    = (__bf16*)alloc((size_t)DM * DM * 2);
  __bf16* KwT    = (__bf16*)alloc((size_t)DM * DM * 2);
  __bf16* VwT    = (__bf16*)alloc((size_t)DM * DM * 2);
  float*  q2     = (float*) alloc((size_t)T_TOK * 4);
  float*  k2     = (float*) alloc(BN * 4);
  float*  u      = (float*) alloc(BN * 4);
  float*  vv     = (float*) alloc((size_t)BATCH * T_TOK * 4);
  float*  Kmat   = (float*) alloc((size_t)BATCH * T_TOK * N_PAR * 4);  // 128 MiB (L2-resident)
  __bf16* pi_bf  = (__bf16*)alloc((size_t)BATCH * T_TOK * N_PAR * 2);  // 64 MiB (L2-resident)

  const int WPB = 4;            // waves per block (proj)
  const dim3 blk(32 * WPB);

  // Weight transpose + bf16 convert (one-shot, 0.5 MB each)
  transpose_w<<<dim3((DM * DM) / (256 * 8)), dim3(256), 0, stream>>>(Qw, QwT);
  transpose_w<<<dim3((DM * DM) / (256 * 8)), dim3(256), 0, stream>>>(Kw, KwT);
  transpose_w<<<dim3((DM * DM) / (256 * 8)), dim3(256), 0, stream>>>(Vw, VwT);

  // Projections (16x64 supertiles: tiles = (M/16) * (DM/64))
  proj_gemm<<<dim3(((T_TOK / 16) * (DM / 64)) / WPB), blk, 0, stream>>>(
      q, QwT, qp, qp_bf, nullptr, T_TOK);
  proj_gemm<<<dim3(((int)(BN / 16) * (DM / 64)) / WPB), blk, 0, stream>>>(
      k, KwT, kp, kp_bf, nullptr, (int)BN);
  proj_gemm<<<dim3(((int)(BN / 16) * (DM / 64)) / WPB), blk, 0, stream>>>(
      v, VwT, nullptr, nullptr, vpT_bf, (int)BN);

  // Row norms
  row_sumsq<<<dim3(T_TOK / 8), dim3(256), 0, stream>>>(qp, q2, T_TOK);
  row_sumsq<<<dim3((int)(BN / 8)), dim3(256), 0, stream>>>(kp, k2, (int)BN);

  // Cost matrix Kmat = -cdist(qp, kp): 128x128 macro tiles
  cost_gemm<<<dim3((T_TOK / MROWS) * (N_PAR / MROWS), BATCH), dim3(256), 0, stream>>>(
      qp_bf, kp_bf, q2, k2, Kmat);

  // Sinkhorn: u = v = 0, then 5x (u-update, v-update)
  hipMemsetAsync(u, 0, BN * 4, stream);
  hipMemsetAsync(vv, 0, (size_t)BATCH * T_TOK * 4, stream);
  for (int it = 0; it < SINK_ITERS; ++it) {
    update_u<<<dim3(N_PAR / 256, BATCH), dim3(256), 0, stream>>>(Kmat, vv, u);
    update_v<<<dim3(T_TOK / 8, BATCH), dim3(256), 0, stream>>>(Kmat, u, vv);
  }

  // pi = exp(Kmat + u + v)  (bf16, computed once, vectorized 8/thread)
  make_pi<<<dim3((unsigned)(((size_t)BATCH * T_TOK * N_PAR) / (256 * 8))), dim3(256), 0, stream>>>(
      Kmat, u, vv, pi_bf);

  // out = pi @ vp: 128x128 macro tiles
  out_gemm<<<dim3((T_TOK / MROWS) * (DM / MROWS), BATCH), dim3(256), 0, stream>>>(
      pi_bf, vpT_bf, out);
}
